// AVIFAdaptiveTransform_55104430408204
// MI455X (gfx1250) — matrix operand order repaired and checked by reading, not compiled, not run on hardware
//
#include <hip/hip_runtime.h>

typedef float  v2f   __attribute__((ext_vector_type(2)));
typedef float  v8f   __attribute__((ext_vector_type(8)));
typedef __bf16 v16bf __attribute__((ext_vector_type(16)));

#define IMG_H 512
#define IMG_W 512
#define NCH   64
#define NB    4

// d_ws layout:
//   [0     .. 8192)  : wpack  __bf16 [2][4][2][32][16]  (w1/w2 A-fragments, per-lane WMMA order)
//   [8192  .. 16384) : bpack  float  [2][4][32][8]      (b1/b2 in C/D-fragment order)
#define WPACK_ELEMS 8192
#define BPACK_OFF   8192

__global__ __launch_bounds__(256) void avif_prepack_kernel(
    const float* __restrict__ w1, const float* __restrict__ b1,
    const float* __restrict__ w2, const float* __restrict__ b2,
    void* __restrict__ ws)
{
    __bf16* wp = (__bf16*)ws;
    float*  bp = (float*)((char*)ws + BPACK_OFF);
    const int tid = threadIdx.x;

    // Weight A-fragments: bf16 16x32 layout, lane m = lane&15, half = lane>>4,
// element s -> k = 16*half + s
    for (int idx = tid; idx < WPACK_ELEMS; idx += 256) {
        const int s    = idx & 15;
        const int lane = (idx >> 4) & 31;
        const int kt   = (idx >> 9) & 1;
        const int mt   = (idx >> 10) & 3;
        const int mtx  = idx >> 12;
        const int nn   = lane & 15, half = lane >> 4;
        const int k    = 16 * half + s;
        const float* Wm = mtx ? w2 : w1;
        wp[idx] = (__bf16)Wm[(16 * mt + nn) * 64 + 32 * kt + k];
    }
    // Bias vectors: C/D layout, element v -> channel 16*mt + v + 8*half
    for (int idx = tid; idx < 2048; idx += 256) {
        const int v    = idx & 7;
        const int lane = (idx >> 3) & 31;
        const int mt   = (idx >> 8) & 3;
        const int mtx  = idx >> 10;
        const int half = lane >> 4;
        const float* Bm = mtx ? b2 : b1;
        bp[idx] = Bm[16 * mt + v + 8 * half];
    }
}

__global__ __launch_bounds__(256) void avif_fused_kernel(
    const float* __restrict__ x,  const float* __restrict__ tw,
    const __bf16* __restrict__ wp, const float* __restrict__ bp,
    float* __restrict__ out)
{
    // LDS: 64KB x/y tile (f32), 16KB transforms, 32KB y (bf16, B-frag order),
    //      32KB per-wave hidden activations (bf16, B-frag order). Total 144KB.
    __shared__ float xs[NCH][256];                       // x tile -> tmp -> y (f32)
    __shared__ float twl[NCH][64];                       // per-channel 8x8 T
    __shared__ __align__(32) __bf16 yb[2][16][32][16];   // [ktile][ntile][lane][slot]
    __shared__ __align__(32) __bf16 hb[8][2][2][32][16]; // [wave][ktile][ntl][lane][slot]

    const int tid  = threadIdx.x;
    const int lane = tid & 31;
    const int wv   = tid >> 5;
    const int nn   = lane & 15;   // n (or m) index within fragment
    const int half = lane >> 4;

    const int tc = blockIdx.x, tr = blockIdx.y, b = blockIdx.z;
    const int row0 = tr * 16, col0 = tc * 16;

    // ---- stage per-channel transforms into LDS (coalesced) ----
    for (int i = tid; i < NCH * 64; i += 256)
        ((float*)twl)[i] = tw[i];

    // ---- load x tile: 64 ch x 16x16 px, float4, coalesced ----
    {
        const float* xb = x + (size_t)b * NCH * IMG_H * IMG_W;
        const int cc = tid >> 6;          // 0..3   (channel within group of 4)
        const int q  = tid & 63;          // 0..63
        const int r  = q >> 2;            // row 0..15
        const int c4 = (q & 3) * 4;       // col 0,4,8,12
        for (int it = 0; it < 16; ++it) {
            const int ch = it * 4 + cc;
            const float4 v = *(const float4*)(xb + (size_t)ch * IMG_H * IMG_W
                                                 + (size_t)(row0 + r) * IMG_W + col0 + c4);
            *(float4*)&xs[ch][r * 16 + c4] = v;
        }
    }

    // ---- prefetch prepacked weight fragments / bias vectors (L2-hot) ----
    v16bf wf1[4][2], wf2[4][2];
    v8f   bias1[4], bias2[4];
    {
        const v16bf* wpv = (const v16bf*)wp;   // [mtx][mt][kt][lane]
        const v8f*   bpv = (const v8f*)bp;     // [mtx][mt][lane]
        #pragma unroll
        for (int mt = 0; mt < 4; ++mt) {
            #pragma unroll
            for (int kt = 0; kt < 2; ++kt) {
                wf1[mt][kt] = wpv[((0 * 4 + mt) * 2 + kt) * 32 + lane];
                wf2[mt][kt] = wpv[((1 * 4 + mt) * 2 + kt) * 32 + lane];
            }
            bias1[mt] = bpv[(0 * 4 + mt) * 32 + lane];
            bias2[mt] = bpv[(1 * 4 + mt) * 32 + lane];
        }
    }
    __syncthreads();

    // =======================================================================
    // Phase 1: per-channel transform Y = T2 * X * T2^T   (fp32 WMMA 16x16x4)
    // T2 = diag(T_c, T_c). Each wave owns 8 channels; all LDS traffic within
    // this phase is same-wave (LDS is in-order per wave -> no barriers).
    // =======================================================================
    for (int ci = 0; ci < 8; ++ci) {
        const int c = wv * 8 + ci;

        // This lane's 8 T2 operand values; identical for the stage-1 A-frag
        // (A[m=nn][k]) and the stage-2 B-frag (T2^T[k][n=nn] = T2[nn][k]).
        // Unconditional LDS load + arithmetic zeroing (no exec-mask branches).
        float tv[8];
        #pragma unroll
        for (int j = 0; j < 8; ++j) {
            const int k = (j >> 1) * 4 + 2 * half + (j & 1);
            const float t = twl[c][(nn & 7) * 8 + (k & 7)];
            tv[j] = (((nn ^ k) & 8) == 0) ? t : 0.0f;
        }

        // Tmp = T2 @ X    (K=16 -> 4 chunks of K=4)
        v8f acc = {};
        #pragma unroll
        for (int i = 0; i < 4; ++i) {
            const int kk = i * 4;
            v2f a, bb;
            a.x  = tv[2 * i + 0];                         // A[m][k], k = kk+2h+v
            a.y  = tv[2 * i + 1];
            bb.x = xs[c][(kk + 2 * half + 0) * 16 + nn];  // B[k][n], k = kk+v+2h
            bb.y = xs[c][(kk + 2 * half + 1) * 16 + nn];
            acc = __builtin_amdgcn_wmma_f32_16x16x4_f32(false, a, false, bb,
                                                        (short)0, acc, false, false);
        }
        // spill Tmp (C/D layout: m = v + 8*half, n = lane&15) back over X
        #pragma unroll
        for (int v = 0; v < 8; ++v)
            xs[c][(v + 8 * half) * 16 + nn] = acc[v];

        // Y = Tmp @ T2^T
        v8f y = {};
        #pragma unroll
        for (int i = 0; i < 4; ++i) {
            const int kk = i * 4;
            v2f a, bb;
            a.x  = xs[c][nn * 16 + kk + 2 * half + 0];    // A = Tmp (contiguous -> b64)
            a.y  = xs[c][nn * 16 + kk + 2 * half + 1];
            bb.x = tv[2 * i + 0];                         // B = T2^T
            bb.y = tv[2 * i + 1];
            y = __builtin_amdgcn_wmma_f32_16x16x4_f32(false, a, false, bb,
                                                      (short)0, y, false, false);
        }

        // store Y: f32 for final multiply; bf16 scattered into B-frag layout
        const int kt = c >> 5, slot = c & 15, lhi = (c >> 4) & 1;
        #pragma unroll
        for (int v = 0; v < 8; ++v) {
            const int m = v + 8 * half;                   // pixel row == gate ntile
            xs[c][m * 16 + nn] = y[v];
            yb[kt][m][nn + 16 * lhi][slot] = (__bf16)y[v];
        }
    }
    __syncthreads();

    // =======================================================================
    // Phase 2: gate = sigmoid(W2 relu(W1 y + b1) + b2); out = y * gate
    // bf16 WMMA 16x16x32, f32 accumulate. M=64(out-ch), K=64(in-ch), N=256(px)
    // =======================================================================
    for (int ntl = 0; ntl < 2; ++ntl) {
        const int nt = 2 * wv + ntl;                      // ntile == pixel row in tile

        // ---- layer 1: h = relu(W1 y + b1) for all 64 hidden channels
        v8f d1[4];
        #pragma unroll
        for (int mt = 0; mt < 4; ++mt) {
            v8f acc = {};
            #pragma unroll
            for (int kt = 0; kt < 2; ++kt) {
                v16bf bF = *(const v16bf*)&yb[kt][nt][lane][0];
                acc = __builtin_amdgcn_wmma_f32_16x16x32_bf16(false, wf1[mt][kt], false, bF,
                                                              (short)0, acc, false, false);
            }
            d1[mt] = acc;
        }
        // bias + relu, scatter into per-wave B-frag-ordered LDS (same-wave, in-order)
        #pragma unroll
        for (int mt = 0; mt < 4; ++mt)
            #pragma unroll
            for (int v = 0; v < 8; ++v) {
                const int ch = 16 * mt + v + 8 * half;
                float h = d1[mt][v] + bias1[mt][v];
                h = h > 0.0f ? h : 0.0f;
                hb[wv][ch >> 5][ntl][nn + 16 * ((ch >> 4) & 1)][ch & 15] = (__bf16)h;
            }

        // ---- layer 2 + sigmoid + multiply + store
        #pragma unroll
        for (int mt = 0; mt < 4; ++mt) {
            v8f acc = {};
            #pragma unroll
            for (int kt = 0; kt < 2; ++kt) {
                v16bf bF = *(const v16bf*)&hb[wv][kt][ntl][lane][0];
                acc = __builtin_amdgcn_wmma_f32_16x16x32_bf16(false, wf2[mt][kt], false, bF,
                                                              (short)0, acc, false, false);
            }
            #pragma unroll
            for (int v = 0; v < 8; ++v) {
                const int oc = 16 * mt + v + 8 * half;
                float g = acc[v] + bias2[mt][v];
                g = 1.0f / (1.0f + __expf(-g));
                const float yv = xs[oc][nt * 16 + nn];
                out[(((size_t)b * NCH + oc) * IMG_H + row0 + nt) * IMG_W + col0 + nn] = yv * g;
            }
        }
    }
}

extern "C" void kernel_launch(void* const* d_in, const int* in_sizes, int n_in,
                              void* d_out, int out_size, void* d_ws, size_t ws_size,
                              hipStream_t stream) {
    const float* x  = (const float*)d_in[0];
    const float* tw = (const float*)d_in[1];
    const float* w1 = (const float*)d_in[2];
    const float* b1 = (const float*)d_in[3];
    const float* w2 = (const float*)d_in[4];
    const float* b2 = (const float*)d_in[5];
    float* out = (float*)d_out;
    (void)in_sizes; (void)n_in; (void)out_size; (void)ws_size;

    // 1) prepack weight fragments + bias vectors into workspace (16 KB)
    avif_prepack_kernel<<<dim3(1), dim3(256), 0, stream>>>(w1, b1, w2, b2, d_ws);

    // 2) fused transform + gate
    const __bf16* wp = (const __bf16*)d_ws;
    const float*  bp = (const float*)((char*)d_ws + BPACK_OFF);
    dim3 grid(IMG_W / 16, IMG_H / 16, NB);   // 32 x 32 x 4 = 4096 workgroups
    avif_fused_kernel<<<grid, dim3(256), 0, stream>>>(x, tw, wp, bp, out);
}
